// DimeNetPlusPlus_2723009266170
// MI455X (gfx1250) — matrix operand order repaired and checked by reading, not compiled, not run on hardware
//
#include <hip/hip_runtime.h>
#include <math.h>

// ---------------- types / constants ----------------
typedef __attribute__((ext_vector_type(16))) __bf16 v16bf;
typedef __attribute__((ext_vector_type(8)))  float  v8f;

#define NRAD 6
#define NSPH 7
#define HID  128
#define INTE 64
#define OEMB 256
#define CUT_INV 0.2f

// fast silu: v * rcp(1+e^-v) — v_rcp_f32 instead of full IEEE divide in the epilogues
__device__ __forceinline__ float siluf(float v){
    return v * __builtin_amdgcn_rcpf(1.0f + __expf(-v));
}

__device__ __forceinline__ __bf16 f2bf(float f){
    unsigned u = __builtin_bit_cast(unsigned, f);
    unsigned r = (u + 0x7FFFu + ((u >> 16) & 1u)) >> 16;
    unsigned short s = (unsigned short)r;
    return __builtin_bit_cast(__bf16, s);
}

// ---------------- spherical bessel helpers ----------------
__device__ double jl_d(int l, double x){
    double jm = sin(x)/x;
    if (l == 0) return jm;
    double jc = sin(x)/(x*x) - cos(x)/x;
    for (int ll = 1; ll < l; ++ll){ double t = (2.0*ll+1.0)/x*jc - jm; jm = jc; jc = t; }
    return jc;
}
__device__ float jl_f(int l, float x){
    float jm = sinf(x)/x;
    if (l == 0) return jm;
    float jc = sinf(x)/(x*x) - cosf(x)/x;
    for (int ll = 1; ll < l; ++ll){ float t = (2.0f*ll+1.0f)/x*jc - jm; jm = jc; jc = t; }
    return jc;
}

// Reproduce _bessel_zeros / _BN / _SPH_C exactly (1 thread, doubles).
__global__ void k_bessel_init(float* Z, float* BN, float* SPHC){
    if (threadIdx.x || blockIdx.x) return;
    const double PI = 3.14159265358979323846;
    double zer[NSPH][NRAD];
    double pts[NRAD + NSPH];
    for (int k = 0; k < NRAD; ++k) zer[0][k] = (k+1)*PI;
    int np_ = NRAD + NSPH - 1;
    for (int i = 0; i < np_; ++i) pts[i] = (i+1)*PI;
    for (int l = 1; l < NSPH; ++l){
        double r[NRAD + NSPH];
        for (int j = 0; j < np_-1; ++j){
            double a = pts[j], b = pts[j+1];
            double fa = jl_d(l, a);
            for (int it = 0; it < 100; ++it){
                double m = 0.5*(a+b);
                if (fa * jl_d(l, m) <= 0.0) b = m;
                else { a = m; fa = jl_d(l, a); }
            }
            r[j] = 0.5*(a+b);
        }
        np_ -= 1;
        for (int j = 0; j < np_; ++j) pts[j] = r[j];
        for (int k = 0; k < NRAD; ++k) zer[l][k] = r[k];
    }
    for (int l = 0; l < NSPH; ++l)
        for (int k = 0; k < NRAD; ++k){
            double zz = zer[l][k];
            Z [l*NRAD+k] = (float)zz;
            double jn = jl_d(l+1, zz);
            BN[l*NRAD+k] = (float)(1.0 / sqrt(0.5 * jn * jn));
        }
    for (int l = 0; l < NSPH; ++l)
        SPHC[l] = (float)sqrt((2.0*l + 1.0) / (4.0*PI));
}

// ---------------- per-edge geometry + radial bases ----------------
__global__ void k_edge(const float* __restrict__ pos, const int* __restrict__ src,
                       const int* __restrict__ dst, const float* __restrict__ freq,
                       const float* __restrict__ Z, const float* __restrict__ BN,
                       float* __restrict__ rbf, float* __restrict__ rbfs, int E){
    int e = blockIdx.x*blockDim.x + threadIdx.x;
    if (e >= E) return;
    int i = dst[e], j = src[e];
    float dx = pos[3*i]-pos[3*j], dy = pos[3*i+1]-pos[3*j+1], dz = pos[3*i+2]-pos[3*j+2];
    float d = sqrtf(dx*dx + dy*dy + dz*dz);
    float x = d * CUT_INV;
    float env = 0.0f;
    if (x < 1.0f){
        const float a = -28.0f, b = 48.0f, c = -21.0f;   // p=6
        float x2 = x*x, x5 = x2*x2*x;
        env = 1.0f/x + a*x5 + b*x5*x + c*x5*x2;
    }
    #pragma unroll
    for (int r = 0; r < NRAD; ++r) rbf[(size_t)e*NRAD + r] = env * sinf(freq[r]*x);
    for (int l = 0; l < NSPH; ++l)
        #pragma unroll
        for (int r = 0; r < NRAD; ++r){
            int idx = l*NRAD + r;
            rbfs[(size_t)e*42 + idx] = env * BN[idx] * jl_f(l, Z[idx]*x);
        }
}

// ---------------- per-triplet Legendre basis ----------------
__global__ void k_cbf(const float* __restrict__ pos, const int* __restrict__ ii,
                      const int* __restrict__ jj, const int* __restrict__ kk,
                      const float* __restrict__ SPHC, float* __restrict__ cbf, int T){
    int t = blockIdx.x*blockDim.x + threadIdx.x;
    if (t >= T) return;
    int i = ii[t], j = jj[t], k = kk[t];
    float jx = pos[3*j]-pos[3*i], jy = pos[3*j+1]-pos[3*i+1], jz = pos[3*j+2]-pos[3*i+2];
    float kx = pos[3*k]-pos[3*j], ky = pos[3*k+1]-pos[3*j+1], kz = pos[3*k+2]-pos[3*j+2];
    float a = jx*kx + jy*ky + jz*kz;
    float cx = jy*kz - jz*ky, cy = jz*kx - jx*kz, cz = jx*ky - jy*kx;
    float b = sqrtf(cx*cx + cy*cy + cz*cz) + 1e-12f;
    float ct = cosf(atan2f(b, a));
    float Pm2 = 1.0f, Pm1 = ct;
    cbf[(size_t)t*NSPH + 0] = SPHC[0]*Pm2;
    cbf[(size_t)t*NSPH + 1] = SPHC[1]*Pm1;
    for (int l = 2; l < NSPH; ++l){
        float P = ((2.0f*l - 1.0f)*ct*Pm1 - (l - 1.0f)*Pm2) / (float)l;
        cbf[(size_t)t*NSPH + l] = SPHC[l]*P;
        Pm2 = Pm1; Pm1 = P;
    }
}

// ---------------- small dense ops ----------------
__global__ void k_zrow(const int* __restrict__ z, const int* __restrict__ eidx,
                       int* __restrict__ out, int E){
    int e = blockIdx.x*blockDim.x + threadIdx.x;
    if (e < E) out[e] = z[eidx[e]];
}

__global__ void k_rbfmlp(const float* __restrict__ rbf, const float* __restrict__ w,
                         const float* __restrict__ b, float* __restrict__ out, int E){
    int idx = blockIdx.x*blockDim.x + threadIdx.x;
    if (idx >= E*HID) return;
    int e = idx >> 7, c = idx & 127;
    float s = b[c];
    #pragma unroll
    for (int r = 0; r < NRAD; ++r) s += rbf[(size_t)e*NRAD + r] * w[r*HID + c];
    out[idx] = siluf(s);
}

__global__ void k_proj8(const float* __restrict__ rbf, const float* __restrict__ w,
                        float* __restrict__ out, int E){
    int idx = blockIdx.x*blockDim.x + threadIdx.x;
    if (idx >= E*8) return;
    int e = idx >> 3, q = idx & 7;
    float s = 0.f;
    #pragma unroll
    for (int r = 0; r < NRAD; ++r) s += rbf[(size_t)e*NRAD + r] * w[r*8 + q];
    out[idx] = s;
}

__global__ void k_mulrbf2(float* __restrict__ xkj, const float* __restrict__ p8,
                          const float* __restrict__ rbf2, int E){
    int idx = blockIdx.x*blockDim.x + threadIdx.x;
    if (idx >= E*HID) return;
    int e = idx >> 7, c = idx & 127;
    float s = 0.f;
    #pragma unroll
    for (int q = 0; q < 8; ++q) s += p8[(size_t)e*8 + q] * rbf2[q*HID + c];
    xkj[idx] *= s;
}

// per-triplet s8 = (rbf_sph[idx_kj]*cbf) @ sbf1   (42 -> 8)
__global__ void k_sbfproj(const float* __restrict__ rbfs, const float* __restrict__ cbf,
                          const int* __restrict__ idx_kj, const float* __restrict__ sbf1,
                          float* __restrict__ s8, int T){
    int t = blockIdx.x*blockDim.x + threadIdx.x;
    if (t >= T) return;
    int kj = idx_kj[t];
    const float* rs = rbfs + (size_t)kj*42;
    const float* cb = cbf + (size_t)t*NSPH;
    float acc[8] = {0,0,0,0,0,0,0,0};
    for (int l = 0; l < NSPH; ++l){
        float cv = cb[l];
        #pragma unroll
        for (int r = 0; r < NRAD; ++r){
            float v = rs[l*NRAD + r] * cv;
            const float* w = sbf1 + (l*NRAD + r)*8;
            #pragma unroll
            for (int q = 0; q < 8; ++q) acc[q] += v * w[q];
        }
    }
    #pragma unroll
    for (int q = 0; q < 8; ++q) s8[(size_t)t*8 + q] = acc[q];
}

// segment-sum over triplets (idx_ji sorted ascending): one wave per edge.
__global__ void k_reduce(const float* __restrict__ s8, const float* __restrict__ sbf2,
                         const float* __restrict__ kjs, const int* __restrict__ idx_kj,
                         const int* __restrict__ idx_ji, float* __restrict__ acc,
                         int E, int T){
    __shared__ float sw[8*INTE];
    for (int i = threadIdx.x; i < 8*INTE; i += blockDim.x) sw[i] = sbf2[i];
    __syncthreads();
    int wId = blockIdx.x*(blockDim.x >> 5) + (threadIdx.x >> 5);
    if (wId >= E) return;
    int lane = threadIdx.x & 31;
    int lo = 0, hi = T;
    while (lo < hi){ int m = (lo+hi) >> 1; if (idx_ji[m] < wId) lo = m+1; else hi = m; }
    int start = lo; hi = T;
    while (lo < hi){ int m = (lo+hi) >> 1; if (idx_ji[m] < wId+1) lo = m+1; else hi = m; }
    int end = lo;
    int n0 = lane, n1 = lane + 32;
    float a0 = 0.f, a1 = 0.f;
    for (int t = start; t < end; ++t){
        int kj = idx_kj[t];
        const float* sp = s8 + (size_t)t*8;
        float x0 = kjs[(size_t)kj*INTE + n0];
        float x1 = kjs[(size_t)kj*INTE + n1];
        float s0 = 0.f, s1 = 0.f;
        #pragma unroll
        for (int q = 0; q < 8; ++q){ float sq = sp[q]; s0 += sq*sw[q*INTE+n0]; s1 += sq*sw[q*INTE+n1]; }
        a0 += x0*s0; a1 += x1*s1;
    }
    acc[(size_t)wId*INTE + n0] = a0;
    acc[(size_t)wId*INTE + n1] = a1;
}

// output block scatter: tnode[i[e]][c] += (rbf@lin_rbf)[e][c] * x[e][c]
__global__ void k_outg(const float* __restrict__ rbf, const float* __restrict__ lr,
                       const float* __restrict__ x, const int* __restrict__ dst,
                       float* __restrict__ tnode, int E){
    int idx = blockIdx.x*blockDim.x + threadIdx.x;
    if (idx >= E*HID) return;
    int e = idx >> 7, c = idx & 127;
    float s = 0.f;
    #pragma unroll
    for (int r = 0; r < NRAD; ++r) s += rbf[(size_t)e*NRAD + r] * lr[r*HID + c];
    atomicAdd(&tnode[(size_t)dst[e]*HID + c], s * x[idx]);
}

// final: out += sum over nodes of t_row . lin
__global__ void k_final(const float* __restrict__ t, const float* __restrict__ lin,
                        float* __restrict__ out, int Nn){
    __shared__ float red[256];
    int node = blockIdx.x*blockDim.x + threadIdx.x;
    float s = 0.f;
    if (node < Nn){
        const float* r = t + (size_t)node*OEMB;
        for (int c = 0; c < OEMB; ++c) s += r[c]*lin[c];
    }
    red[threadIdx.x] = s; __syncthreads();
    for (int st = 128; st > 0; st >>= 1){
        if (threadIdx.x < st) red[threadIdx.x] += red[threadIdx.x + st];
        __syncthreads();
    }
    if (threadIdx.x == 0) atomicAdd(out, red[0]);
}

// ---------------- weight pack: f32 [K x N] -> bf16 fragment order [K/32][N][32] ----------------
__global__ void k_pack(const float* __restrict__ in, __bf16* __restrict__ out, int K, int N){
    int idx = blockIdx.x*blockDim.x + threadIdx.x;
    if (idx >= K*N) return;
    int k = idx / N, n = idx - k*N;
    out[((size_t)(k >> 5)*N + n)*32 + (k & 31)] = f2bf(in[idx]);
}

// ---------------- generic WMMA GEMM ----------------
// C[M,N](ldc) = epilogue( A[M,K](lda, optional row gather) @ Bp[packed bf16] )
// flags: 1=beta (accumulate from C), 2=+bias, 4=silu, 8=+add (ldc layout)
// Each wave computes a 16x32 output tile: 1 A-fragment x 2 B-fragments -> 2 WMMAs/k-step.
__global__ void k_gemm(const float* __restrict__ A, int lda, const int* __restrict__ rowidx,
                       const __bf16* __restrict__ Bp, const float* __restrict__ bias,
                       const float* __restrict__ add, float* __restrict__ C, int ldc,
                       int M, int N, int K, int flags){
    const int tilesP = N >> 5;                 // pairs of 16-wide tiles
    int total = ((M + 15) >> 4) * tilesP;
    int wid = blockIdx.x*(blockDim.x >> 5) + (threadIdx.x >> 5);
    if (wid >= total) return;
    int tm = wid / tilesP, tp = wid - tm*tilesP;
    int lane = threadIdx.x & 31;
    int lh = lane >> 4, l15 = lane & 15;

    // clamp out-of-range rows to valid memory; garbage only lands in D rows never stored
    int arow = tm*16 + l15; if (arow >= M) arow = M - 1;
    long asrc = rowidx ? (long)rowidx[arow] : (long)arow;
    const float* Arow = A + (size_t)asrc * lda + 16*lh;

    int row8 = tm*16 + 8*lh;
    int col0 = tp*32 + l15;
    int col1 = col0 + 16;

    v8f c0, c1;
    if (flags & 1){
        #pragma unroll
        for (int r = 0; r < 8; ++r){
            int rr = row8 + r; if (rr >= M) rr = 0;
            c0[r] = C[(size_t)rr*ldc + col0];
            c1[r] = C[(size_t)rr*ldc + col1];
        }
    } else {
        #pragma unroll
        for (int r = 0; r < 8; ++r){ c0[r] = 0.f; c1[r] = 0.f; }
    }

    const __bf16* bp0 = Bp + (size_t)col0*32 + 16*lh;
    const __bf16* bp1 = Bp + (size_t)col1*32 + 16*lh;
    for (int k0 = 0; k0 < K; k0 += 32){
        float4 a0 = *(const float4*)(Arow + k0);
        float4 a1 = *(const float4*)(Arow + k0 + 4);
        float4 a2 = *(const float4*)(Arow + k0 + 8);
        float4 a3 = *(const float4*)(Arow + k0 + 12);
        v16bf af;
        af[0] = f2bf(a0.x); af[1] = f2bf(a0.y); af[2]  = f2bf(a0.z); af[3]  = f2bf(a0.w);
        af[4] = f2bf(a1.x); af[5] = f2bf(a1.y); af[6]  = f2bf(a1.z); af[7]  = f2bf(a1.w);
        af[8] = f2bf(a2.x); af[9] = f2bf(a2.y); af[10] = f2bf(a2.z); af[11] = f2bf(a2.w);
        af[12]= f2bf(a3.x); af[13]= f2bf(a3.y); af[14] = f2bf(a3.z); af[15] = f2bf(a3.w);
        size_t kb = (size_t)(k0 >> 5) * N * 32;
        v16bf b0 = *(const v16bf*)(bp0 + kb);
        v16bf b1 = *(const v16bf*)(bp1 + kb);
        c0 = __builtin_amdgcn_wmma_f32_16x16x32_bf16(false, af, false, b0, (short)0, c0, false, false);
        c1 = __builtin_amdgcn_wmma_f32_16x16x32_bf16(false, af, false, b1, (short)0, c1, false, false);
    }

    #pragma unroll
    for (int r = 0; r < 8; ++r){
        int rr = row8 + r;
        if (rr >= M) continue;
        float v0 = c0[r], v1 = c1[r];
        if (flags & 2){ v0 += bias[col0]; v1 += bias[col1]; }
        if (flags & 4){ v0 = siluf(v0);   v1 = siluf(v1); }
        if (flags & 8){ v0 += add[(size_t)rr*ldc + col0]; v1 += add[(size_t)rr*ldc + col1]; }
        C[(size_t)rr*ldc + col0] = v0;
        C[(size_t)rr*ldc + col1] = v1;
    }
}

// ---------------- host side ----------------
static inline int cdiv(long a, long b){ return (int)((a + b - 1)/b); }

static void gemm(hipStream_t s, const float* A, int lda, const int* ridx, const __bf16* Bp,
                 const float* bias, const float* add, float* C, int ldc,
                 int M, int N, int K, int flags){
    int total = ((M + 15)/16) * (N/32);
    int grid = cdiv(total, 8);                 // 8 waves / 256-thread block
    hipLaunchKernelGGL(k_gemm, dim3(grid), dim3(256), 0, s,
                       A, lda, ridx, Bp, bias, add, C, ldc, M, N, K, flags);
}

extern "C" void kernel_launch(void* const* d_in, const int* in_sizes, int n_in,
                              void* d_out, int out_size, void* d_ws, size_t ws_size,
                              hipStream_t stream){
    if (n_in < 156) return;
    const int*   z      = (const int*)  d_in[0];
    const float* pos    = (const float*)d_in[1];
    const int*   esrc   = (const int*)  d_in[2];
    const int*   edst   = (const int*)  d_in[3];
    const int*   idx_i  = (const int*)  d_in[4];
    const int*   idx_j  = (const int*)  d_in[5];
    const int*   idx_k  = (const int*)  d_in[6];
    const int*   idx_kj = (const int*)  d_in[7];
    const int*   idx_ji = (const int*)  d_in[8];
    const int Nn = in_sizes[0];
    const int E  = in_sizes[2];
    const int T  = in_sizes[4];
    auto F = [&](int i)->const float*{ return (const float*)d_in[i]; };
    // params flatten (top-level insertion order; params pytree: sorted dict keys, tuples in order)
    //  9 emb.lin_b  10 emb.lin_w  11 emb.rbf_b  12 emb.rbf_w  13 emb.table  14 freq
    //  15+24b int block: +0..3 after0(w1,b1,w2,b2) +4..7 after1 +8..11 before0
    //    +12 down +13 ji_b +14 ji_w +15 kj_b +16 kj_w +17 lin_b +18 lin_w
    //    +19 rbf1 +20 rbf2 +21 sbf1 +22 sbf2 +23 up
    //  111+9o out block: +0 lin +1 lin_rbf +2 lin_up +3 l0w +4 l0b +5 l1w +6 l1b +7 l2w +8 l2b

    // workspace carve-up (256B aligned bump allocator)
    char* wptr = (char*)d_ws;
    auto allocb = [&](size_t bytes)->void*{
        void* p = (void*)wptr;
        wptr += ((bytes + 255)/256)*256;
        return p;
    };
    auto alloc = [&](size_t nf)->float*{ return (float*)allocb(nf*sizeof(float)); };

    float* Zc   = alloc(64);
    float* BNc  = alloc(64);
    float* SPHC = alloc(16);
    float* rbf  = alloc((size_t)E*NRAD);
    float* rbfs = alloc((size_t)E*42);
    float* p8   = alloc((size_t)E*8);
    float* cbf  = alloc((size_t)T*NSPH);
    float* s8   = alloc((size_t)T*8);
    int*   zi   = (int*)alloc(E);
    int*   zj   = (int*)alloc(E);
    float* P0   = alloc((size_t)E*HID);
    float* P1   = alloc((size_t)E*HID);
    float* P2   = alloc((size_t)E*HID);
    float* P3   = alloc((size_t)E*HID);
    float* C64  = alloc((size_t)E*HID);        // kjs (E*64) + acc (E*64)
    float* kjs  = C64;
    float* acc  = C64 + (size_t)E*INTE;
    float* tn   = alloc((size_t)Nn*HID);
    float* ta   = alloc((size_t)Nn*OEMB);
    float* tb   = alloc((size_t)Nn*OEMB);

    // pack all GEMM weights to bf16 fragment order
    auto packW = [&](int leaf, int K, int N)->const __bf16*{
        __bf16* dst = (__bf16*)allocb((size_t)K*N*sizeof(__bf16));
        hipLaunchKernelGGL(k_pack, dim3(cdiv((long)K*N,256)), dim3(256), 0, stream,
                           F(leaf), dst, K, N);
        return dst;
    };
    const __bf16* embW = packW(10, 3*HID, HID);          // 384x128 (3 K-segments of 128)
    const __bf16 *jiW[4], *kjW[4], *dnW[4], *upW[4], *b0w1[4], *b0w2[4], *linW[4];
    const __bf16 *a0w1[4], *a0w2[4], *a1w1[4], *a1w2[4];
    for (int b = 0; b < 4; ++b){
        int ib = 15 + b*24;
        a0w1[b] = packW(ib+0,  HID, HID);
        a0w2[b] = packW(ib+2,  HID, HID);
        a1w1[b] = packW(ib+4,  HID, HID);
        a1w2[b] = packW(ib+6,  HID, HID);
        b0w1[b] = packW(ib+8,  HID, HID);
        b0w2[b] = packW(ib+10, HID, HID);
        dnW[b]  = packW(ib+12, HID, INTE);
        jiW[b]  = packW(ib+14, HID, HID);
        kjW[b]  = packW(ib+16, HID, HID);
        linW[b] = packW(ib+18, HID, HID);
        upW[b]  = packW(ib+23, INTE, HID);
    }
    const __bf16 *lupW[5], *lW[5][3];
    for (int o = 0; o < 5; ++o){
        int ob = 111 + o*9;
        lupW[o] = packW(ob+2, HID, OEMB);
        for (int li = 0; li < 3; ++li) lW[o][li] = packW(ob+3+2*li, OEMB, OEMB);
    }

    float* outp = (float*)d_out;
    hipMemsetAsync(outp, 0, sizeof(float), stream);

    // basis constants + edge/triplet bases
    hipLaunchKernelGGL(k_bessel_init, dim3(1), dim3(1), 0, stream, Zc, BNc, SPHC);
    hipLaunchKernelGGL(k_edge, dim3(cdiv(E,256)), dim3(256), 0, stream,
                       pos, esrc, edst, F(14), Zc, BNc, rbf, rbfs, E);
    hipLaunchKernelGGL(k_cbf, dim3(cdiv(T,256)), dim3(256), 0, stream,
                       pos, idx_i, idx_j, idx_k, SPHC, cbf, T);
    hipLaunchKernelGGL(k_zrow, dim3(cdiv(E,256)), dim3(256), 0, stream, z, edst, zi, E);
    hipLaunchKernelGGL(k_zrow, dim3(cdiv(E,256)), dim3(256), 0, stream, z, esrc, zj, E);

    // embedding: rbf_e = silu(rbf@rbf_w+b);  x = silu([h_i,h_j,rbf_e]@lin_w + lin_b)
    hipLaunchKernelGGL(k_rbfmlp, dim3(cdiv((long)E*HID,256)), dim3(256), 0, stream,
                       rbf, F(12), F(11), P1, E);
    const size_t segOff = (size_t)(HID/32)*HID*32;       // one 128-row K segment in packed layout
    gemm(stream, F(13), HID, zi,      embW,            nullptr, nullptr, P0, HID, E, HID, HID, 0);
    gemm(stream, F(13), HID, zj,      embW +   segOff, nullptr, nullptr, P0, HID, E, HID, HID, 1);
    gemm(stream, P1,    HID, nullptr, embW + 2*segOff, F(9),    nullptr, P0, HID, E, HID, HID, 1|2|4);

    float* x = P0; float* bufA = P1; float* bufB = P2; float* bufC = P3;

    auto out_block = [&](int o, const float* xcur){
        int ob = 111 + o*9;
        hipMemsetAsync(tn, 0, (size_t)Nn*HID*sizeof(float), stream);
        hipLaunchKernelGGL(k_outg, dim3(cdiv((long)E*HID,256)), dim3(256), 0, stream,
                           rbf, F(ob+1), xcur, edst, tn, E);
        gemm(stream, tn, HID, nullptr, lupW[o], nullptr, nullptr, ta, OEMB, Nn, OEMB, HID, 0);
        float* cur = ta; float* nxt = tb;
        for (int li = 0; li < 3; ++li){
            gemm(stream, cur, OEMB, nullptr, lW[o][li], F(ob+4+2*li), nullptr,
                 nxt, OEMB, Nn, OEMB, OEMB, 2|4);
            float* t2 = cur; cur = nxt; nxt = t2;
        }
        hipLaunchKernelGGL(k_final, dim3(cdiv(Nn,256)), dim3(256), 0, stream,
                           cur, F(ob+0), outp, Nn);
    };

    out_block(0, x);

    for (int b = 0; b < 4; ++b){
        int ib = 15 + b*24;
        // x_ji, x_kj
        gemm(stream, x, HID, nullptr, jiW[b], F(ib+13), nullptr, bufA, HID, E, HID, HID, 2|4);
        gemm(stream, x, HID, nullptr, kjW[b], F(ib+15), nullptr, bufB, HID, E, HID, HID, 2|4);
        // x_kj *= rbf@rbf1@rbf2
        hipLaunchKernelGGL(k_proj8, dim3(cdiv((long)E*8,256)), dim3(256), 0, stream,
                           rbf, F(ib+19), p8, E);
        hipLaunchKernelGGL(k_mulrbf2, dim3(cdiv((long)E*HID,256)), dim3(256), 0, stream,
                           bufB, p8, F(ib+20), E);
        // down-project
        gemm(stream, bufB, HID, nullptr, dnW[b], nullptr, nullptr, kjs, INTE, E, INTE, HID, 4);
        // triplet path
        hipLaunchKernelGGL(k_sbfproj, dim3(cdiv(T,256)), dim3(256), 0, stream,
                           rbfs, cbf, idx_kj, F(ib+21), s8, T);
        hipLaunchKernelGGL(k_reduce, dim3(cdiv(E,8)), dim3(256), 0, stream,
                           s8, F(ib+22), kjs, idx_kj, idx_ji, acc, E, T);
        // up-project + x_ji
        gemm(stream, acc, INTE, nullptr, upW[b], nullptr, bufA, bufB, HID, E, HID, INTE, 4|8);
        // before residual (1 layer)
        gemm(stream, bufB, HID, nullptr, b0w1[b], F(ib+9),  nullptr, bufA, HID, E, HID, HID, 2|4);
        gemm(stream, bufA, HID, nullptr, b0w2[b], F(ib+11), bufB,    bufB, HID, E, HID, HID, 2|4|8);
        // lin + x
        gemm(stream, bufB, HID, nullptr, linW[b], F(ib+17), x, bufC, HID, E, HID, HID, 2|4|8);
        { float* old = x; x = bufC; bufC = old; }
        // after residuals (2 layers)
        gemm(stream, x,    HID, nullptr, a0w1[b], F(ib+1), nullptr, bufA, HID, E, HID, HID, 2|4);
        gemm(stream, bufA, HID, nullptr, a0w2[b], F(ib+3), x,       x,    HID, E, HID, HID, 2|4|8);
        gemm(stream, x,    HID, nullptr, a1w1[b], F(ib+5), nullptr, bufA, HID, E, HID, HID, 2|4);
        gemm(stream, bufA, HID, nullptr, a1w2[b], F(ib+7), x,       x,    HID, E, HID, HID, 2|4|8);
        out_block(b+1, x);
    }
}
